// SimplicialMessagePassingBlock_58222576664706
// MI455X (gfx1250) — matrix operand order repaired and checked by reading, not compiled, not run on hardware
//
#include <hip/hip_runtime.h>

// SimplicialMessagePassingBlock on MI455X (gfx1250, wave32):
//   h   = x @ W.T + b          [16384, 256]   (kernel 1, bf16 WMMA, stored transposed in d_ws)
//   out = adj @ h              [16384, 256]   (kernel 2, bf16 WMMA, adj streamed NT from HBM)
//
// Roofline: adj = 1 GiB fp32 -> 46us at 23.3 TB/s. 137 GFLOP needs ~3 PFLOPS to stay
// memory bound -> only the bf16 V_WMMA_F32_16X16X32_BF16 path gets there. fp32 WMMA
// (16x16x4) would be ~5x compute bound. So: convert adj/h to bf16 in-register, f32 accum.

#define N_NODES 16384
#define D_FEAT  256

typedef __attribute__((ext_vector_type(16))) __bf16 v16bf;
typedef __attribute__((ext_vector_type(8)))  float  v8f;
typedef __attribute__((ext_vector_type(4)))  float  f32x4;

// ---- A-fragment (16x32 bf16, contiguous-K order used by this file) from fp32 row-major memory ----
// lane<16 : row M=lane,    K = {k0..k0+15} in halves 0..15
// lane>=16: row M=lane-16, K = {k0+16..k0+31}
// caller passes p = row_base + k0 + (lane>>4)*16
__device__ __forceinline__ v16bf frag_a_f32_nt(const float* p) {
    f32x4 a0 = __builtin_nontemporal_load((const f32x4*)(p + 0));
    f32x4 a1 = __builtin_nontemporal_load((const f32x4*)(p + 4));
    f32x4 a2 = __builtin_nontemporal_load((const f32x4*)(p + 8));
    f32x4 a3 = __builtin_nontemporal_load((const f32x4*)(p + 12));
    v16bf r;
#pragma unroll
    for (int i = 0; i < 4; ++i) {
        r[i]      = (__bf16)a0[i];
        r[4 + i]  = (__bf16)a1[i];
        r[8 + i]  = (__bf16)a2[i];
        r[12 + i] = (__bf16)a3[i];
    }
    return r;
}

__device__ __forceinline__ v16bf frag_a_f32(const float* p) {
    f32x4 a0 = *(const f32x4*)(p + 0);
    f32x4 a1 = *(const f32x4*)(p + 4);
    f32x4 a2 = *(const f32x4*)(p + 8);
    f32x4 a3 = *(const f32x4*)(p + 12);
    v16bf r;
#pragma unroll
    for (int i = 0; i < 4; ++i) {
        r[i]      = (__bf16)a0[i];
        r[4 + i]  = (__bf16)a1[i];
        r[8 + i]  = (__bf16)a2[i];
        r[12 + i] = (__bf16)a3[i];
    }
    return r;
}

// ---- B-fragment (32x16 bf16): lane<16 holds K=k0..k0+15 of col N=lane,
//      lane>=16 holds K=k0+16..k0+31 of col N=lane-16. 16 contiguous values.
// caller passes p = col_base + k0 + (lane>>4)*16 (fp32 source, converted)
__device__ __forceinline__ v16bf frag_b_f32(const float* p) {
    f32x4 a0 = *(const f32x4*)(p + 0);
    f32x4 a1 = *(const f32x4*)(p + 4);
    f32x4 a2 = *(const f32x4*)(p + 8);
    f32x4 a3 = *(const f32x4*)(p + 12);
    v16bf r;
#pragma unroll
    for (int i = 0; i < 4; ++i) {
        r[i]      = (__bf16)a0[i];
        r[4 + i]  = (__bf16)a1[i];
        r[8 + i]  = (__bf16)a2[i];
        r[12 + i] = (__bf16)a3[i];
    }
    return r;
}

// =====================================================================
// Kernel 1: hT[e][n] = bf16( sum_d x[n][d] * W[e][d] + b[e] )
// One wave per 16(n) x 16(e) tile; K=256 in 8 steps of 32.
// B tile is W^T[d][e] = W[e][d]: row-major W rows ARE the B columns -> contiguous.
// =====================================================================
__global__ __launch_bounds__(128) void smp_proj_kernel(
    const float* __restrict__ x, const float* __restrict__ W,
    const float* __restrict__ b, __bf16* __restrict__ hT) {
    const int wave  = (blockIdx.x * blockDim.x + threadIdx.x) >> 5;
    const int lane  = threadIdx.x & 31;
    const int mTile = wave >> 4;    // 0..1023 node tile
    const int eTile = wave & 15;    // 0..15 feature tile
    const int hi    = lane >> 4;
    const int nrow  = mTile * 16 + (lane & 15);  // node row for A
    const int erow  = eTile * 16 + (lane & 15);  // feature row of W = B column

    v8f acc = {};
    const float* xrow = x + (size_t)nrow * D_FEAT;
    const float* wrow = W + (size_t)erow * D_FEAT;
#pragma unroll
    for (int k0 = 0; k0 < D_FEAT; k0 += 32) {
        v16bf a  = frag_a_f32(xrow + k0 + hi * 16);
        v16bf bf = frag_b_f32(wrow + k0 + hi * 16);
        acc = __builtin_amdgcn_wmma_f32_16x16x32_bf16(
            false, a, false, bf, (short)0, acc, false, false);
    }
    const float bias = b[erow];
#pragma unroll
    for (int r = 0; r < 8; ++r) {
        const int n = mTile * 16 + r + hi * 8;     // D layout: M = r + 8*hi
        hT[(size_t)erow * N_NODES + n] = (__bf16)(acc[r] + bias);
    }
}

// =====================================================================
// Kernel 2: out = adj @ h.  Wave computes 16 rows x 128 cols (8 N-tiles);
// two waves per m-tile (N split) share a block so adj lines are fetched once.
// adj loaded non-temporally (streamed 1 GiB must not evict 8 MB hT from L2).
// =====================================================================
__global__ __launch_bounds__(256) void smp_agg_kernel(
    const float* __restrict__ adj, const __bf16* __restrict__ hT,
    float* __restrict__ out) {
    const int wib   = threadIdx.x >> 5;                 // wave in block: 0..7
    const int lane  = threadIdx.x & 31;
    const int mTile = blockIdx.x * 4 + (wib >> 1);      // 0..1023
    const int nHalf = wib & 1;                          // 0..1 (8 N-tiles each)
    const int hi    = lane >> 4;
    const int mrow  = mTile * 16 + (lane & 15);

    v8f acc[8];
#pragma unroll
    for (int j = 0; j < 8; ++j) acc[j] = (v8f){};

    const float* arow = adj + (size_t)mrow * N_NODES;

    for (int k0 = 0; k0 < N_NODES; k0 += 32) {
        const v16bf a = frag_a_f32_nt(arow + k0 + hi * 16);
#pragma unroll
        for (int j = 0; j < 8; ++j) {
            const int ncol = (nHalf * 8 + j) * 16 + (lane & 15);  // B column
            const v16bf bf = *(const v16bf*)(hT + (size_t)ncol * N_NODES + k0 + hi * 16);
            acc[j] = __builtin_amdgcn_wmma_f32_16x16x32_bf16(
                false, a, false, bf, (short)0, acc[j], false, false);
        }
    }

#pragma unroll
    for (int j = 0; j < 8; ++j) {
        const int ncol = (nHalf * 8 + j) * 16 + (lane & 15);
#pragma unroll
        for (int r = 0; r < 8; ++r) {
            const int m = mTile * 16 + r + hi * 8;      // D layout: M = r + 8*hi
            __builtin_nontemporal_store(acc[j][r], out + (size_t)m * D_FEAT + ncol);
        }
    }
}

extern "C" void kernel_launch(void* const* d_in, const int* in_sizes, int n_in,
                              void* d_out, int out_size, void* d_ws, size_t ws_size,
                              hipStream_t stream) {
    const float* x   = (const float*)d_in[0];   // [16384, 256]
    const float* adj = (const float*)d_in[1];   // [16384, 16384]
    const float* W   = (const float*)d_in[2];   // [256, 256]
    const float* b   = (const float*)d_in[3];   // [256]
    float*       out = (float*)d_out;           // [16384, 256]
    __bf16*      hT  = (__bf16*)d_ws;           // [256, 16384] bf16 = 8 MB scratch

    // Kernel 1: 16384 wave-tiles (1024 m-tiles x 16 e-tiles), 4 waves/block.
    smp_proj_kernel<<<4096, 128, 0, stream>>>(x, W, b, hT);
    // Kernel 2: 2048 waves (1024 m-tiles x 2 N-halves), 8 waves/block.
    smp_agg_kernel<<<256, 256, 0, stream>>>(adj, hT, out);
}
